// DecoderLSTM_88880053223690
// MI455X (gfx1250) — matrix operand order repaired and checked by reading, not compiled, not run on hardware
//
#include <hip/hip_runtime.h>
#include <hip/hip_bf16.h>

typedef __attribute__((ext_vector_type(16))) _Float16 v16h;
typedef __attribute__((ext_vector_type(8)))  _Float16 v8h;
typedef __attribute__((ext_vector_type(8)))  float    v8f;
typedef __attribute__((ext_vector_type(4)))  unsigned int u32x4;
typedef __attribute__((ext_vector_type(8)))  int      i32x8;
typedef __attribute__((ext_vector_type(4)))  int      i32x4;

#define BQ 256   // batch
#define HQ 256   // hidden
#define GQ 1024  // 4*H gates

// ---------------------------------------------------------------------------
// fp32 -> f16 pack with K-padding (used for activations AND weights).
__global__ void k_padpack(const float* __restrict__ src, _Float16* __restrict__ dst,
                          int n, int din, int dpad)
{
    int idx = blockIdx.x * blockDim.x + threadIdx.x;
    if (idx >= n) return;
    int k = idx % dpad;
    int r = idx / dpad;
    dst[idx] = (k < din) ? (_Float16)src[(size_t)r * din + k] : (_Float16)0.f;
}

// conv weights (O,I,K) fp32 -> (K,O,I) f16 for WMMA conv
__global__ void k_convw_pack(const float* __restrict__ w, _Float16* __restrict__ dst,
                             int Cout, int Cin, int ks, int n)
{
    int idx = blockIdx.x * blockDim.x + threadIdx.x;
    if (idx >= n) return;
    int dk  = idx / (Cout * Cin);
    int rem = idx % (Cout * Cin);
    int o   = rem / Cin;
    int ci  = rem % Cin;
    dst[idx] = (_Float16)w[((size_t)o * Cin + ci) * ks + dk];
}

// ---------------------------------------------------------------------------
__global__ void k_init_state(const float* __restrict__ h0, const float* __restrict__ c0,
                             _Float16* __restrict__ h16, float* __restrict__ cws)
{
    int idx = blockIdx.x * blockDim.x + threadIdx.x;
    if (idx >= 2 * BQ * HQ) return;
    h16[idx] = (_Float16)h0[idx];
    cws[idx] = c0[idx];
}

// Time-gather (B,Tin,C) -> (B,Tout,C), src_t = (t*mul)/div
__global__ void k_gather_f32(const float* __restrict__ src, float* __restrict__ dst,
                             int Tin, int Tout, int C, int mul, int dv, int n)
{
    int idx = blockIdx.x * blockDim.x + threadIdx.x;
    if (idx >= n) return;
    int c = idx % C;
    int t = (idx / C) % Tout;
    int b = idx / (C * Tout);
    int st = (t * mul) / dv;
    dst[idx] = src[((size_t)b * Tin + st) * C + c];
}

__global__ void k_build_in5(const float* __restrict__ x1, const float* __restrict__ x3,
                            float* __restrict__ dst, int n)
{
    int idx = blockIdx.x * blockDim.x + threadIdx.x;
    if (idx >= n) return;
    int c = idx % 3;
    int t = (idx / 3) % 96;
    int b = idx / (3 * 96);
    float v;
    if (c == 0) v = x3[((size_t)b * 96 + (95 - t)) * 2 + 0];
    else        v = x1[((size_t)b * 96 + t) * 2 + (c - 1)];
    dst[idx] = v;
}

// ---------------------------------------------------------------------------
// Tensor Data Mover: async 2-D tile load global->LDS (CDNA5 TDM, D# per ISA §8).
// rows x rowlen elements (2B each), row stride rowstride_el; LDS gets +16B pad
// after every row (pad_amount=4 dwords, pad_interval = row length, pow2).
#if __has_builtin(__builtin_amdgcn_tensor_load_to_lds)
#define HAVE_TDM 1
__device__ inline void tdm_load_2d(const void* gsrc, unsigned lds_off,
                                   int rowlen_el, int nrows, int rowstride_el,
                                   int pad_code)
{
    unsigned long long ga = (unsigned long long)(size_t)gsrc;
    u32x4 g0 = {};
    g0[0] = 1u;                                   // count=1 (valid descriptor)
    g0[1] = lds_off;                              // lds_addr (bytes)
    g0[2] = (unsigned)(ga & 0xFFFFFFFFu);         // global_addr[31:0]
    g0[3] = (unsigned)((ga >> 32) & 0x1FFFFFFu)   // global_addr[56:32]
          | (2u << 30);                           // type = 2 ("image")
    i32x8 g1 = {};
    g1[0] = (1 << 16)                             // data_size = 2 bytes
          | (1 << 20)                             // pad_enable
          | (pad_code << 22)                      // pad_interval (row len, pow2 dwords)
          | (3 << 25);                            // pad_amount = 4 dwords (8 halves)
    g1[1] = (rowlen_el & 0xFFFF) << 16;           // tensor_dim0[15:0]
    g1[2] = ((rowlen_el >> 16) & 0xFFFF)          // tensor_dim0[31:16]
          | ((nrows & 0xFFFF) << 16);             // tensor_dim1[15:0]
    g1[3] = ((nrows >> 16) & 0xFFFF)              // tensor_dim1[31:16]
          | ((rowlen_el & 0xFFFF) << 16);         // tile_dim0
    g1[4] = (nrows & 0xFFFF);                     // tile_dim1 (tile_dim2 = 0)
    g1[5] = rowstride_el;                         // tensor_dim0_stride[31:0]
    g1[6] = 0;
    g1[7] = 0;
    i32x4 z4 = {};
#if __clang_major__ >= 23
    i32x8 z8 = {};
    __builtin_amdgcn_tensor_load_to_lds(g0, g1, z4, z4, z8, 0);
#else
    __builtin_amdgcn_tensor_load_to_lds(g0, g1, z4, z4, 0);
#endif
}
#else
#define HAVE_TDM 0
#endif

// ---------------------------------------------------------------------------
// LSTM timestep GEMM with TDM/LDS-staged A operand.
//   gates[dir] = [X_t | h] @ [Wih | Whh]^T + bias
// grid = (16 mt, 2 nslice, 2 dir), block = 256 (8 waves).
// Wave 0 issues two TDM tile loads (X rows, h rows) into bank-conflict-padded
// LDS regions and waits TENSORcnt; each wave then computes 4 N-tiles with the
// A fragment read via ds_load_b128 and 4 B fragments from global (L2-resident).
__global__ __launch_bounds__(256) void k_lstm_gemm(
    const _Float16* __restrict__ X16, int ldx, int dpad,
    const _Float16* __restrict__ Wih, const _Float16* __restrict__ Whh,
    const float* __restrict__ bias,
    const _Float16* __restrict__ h16,
    float* __restrict__ gates, int T, int s)
{
    __shared__ __align__(16) _Float16 sX[16 * (512 + 8)];
    __shared__ __align__(16) _Float16 sH[16 * (256 + 8)];
    const int mt  = blockIdx.x;
    const int nsl = blockIdx.y;
    const int dir = blockIdx.z;
    const int t   = dir ? (T - 1 - s) : s;
    const int tid = threadIdx.x;
    const int KPX = dpad + 8;           // padded LDS row strides (halves)
    const int KPH = HQ + 8;

#if HAVE_TDM
    if (tid < 32) {                      // wave 0 issues the tile DMAs
        const _Float16* gx = X16 + (size_t)(mt * 16) * ldx + (size_t)t * dpad;
        const _Float16* gh = h16 + (size_t)dir * BQ * HQ + (size_t)(mt * 16) * HQ;
        tdm_load_2d(gx, (unsigned)(size_t)(void*)sX, dpad, 16, ldx,
                    dpad == 512 ? 7 : 3);           // 256-dw / 16-dw rows
        tdm_load_2d(gh, (unsigned)(size_t)(void*)sH, HQ, 16, HQ, 6); // 128-dw rows
        __builtin_amdgcn_s_wait_tensorcnt(0);
    }
#else
    {   // cooperative fallback fill
        const int nchx = (16 * dpad) >> 3;
        for (int c = tid; c < nchx; c += 256) {
            int e = c << 3, r = e / dpad, k = e % dpad;
            *(v8h*)(sX + r * KPX + k) =
                *(const v8h*)(X16 + (size_t)(mt * 16 + r) * ldx + (size_t)t * dpad + k);
        }
        const int nchh = (16 * HQ) >> 3;
        for (int c = tid; c < nchh; c += 256) {
            int e = c << 3, r = e >> 8, k = e & 255;
            *(v8h*)(sH + r * KPH + k) =
                *(const v8h*)(h16 + (size_t)dir * BQ * HQ + (size_t)(mt * 16 + r) * HQ + k);
        }
    }
#endif
    __syncthreads();

    const int wave = tid >> 5, lane = tid & 31;
    const int ml = lane & 15, hi = lane >> 4;
    const int nt0 = nsl * 32 + wave * 4;         // 4 consecutive N-tiles
    const _Float16* sXr = sX + ml * KPX + hi * 8;
    const _Float16* sHr = sH + ml * KPH + hi * 8;

    v8f acc[4] = {{}, {}, {}, {}};

    // ---- K over input-projection part ----
    {
        const _Float16* Bb0 = Wih + ((size_t)dir * GQ + nt0 * 16 + ml) * dpad + hi * 8;
        __builtin_prefetch(Bb0, 0, 1);
        for (int kk = 0; kk < dpad; kk += 32) {
            v8h a0 = *(const v8h*)(sXr + kk);
            v8h a1 = *(const v8h*)(sXr + kk + 16);
            v16h Af;
#pragma unroll
            for (int i = 0; i < 8; i++) { Af[i] = a0[i]; Af[i + 8] = a1[i]; }
#pragma unroll
            for (int j = 0; j < 4; j++) {
                const _Float16* Bb = Bb0 + (size_t)j * 16 * dpad + kk;
                v8h b0 = *(const v8h*)(Bb);
                v8h b1 = *(const v8h*)(Bb + 16);
                v16h Bf;
#pragma unroll
                for (int i = 0; i < 8; i++) { Bf[i] = b0[i]; Bf[i + 8] = b1[i]; }
                acc[j] = __builtin_amdgcn_wmma_f32_16x16x32_f16(
                    false, Af, false, Bf, (short)0, acc[j], false, false);
            }
        }
    }
    // ---- K over recurrent part ----
    {
        const _Float16* Bb0 = Whh + ((size_t)dir * GQ + nt0 * 16 + ml) * HQ + hi * 8;
        __builtin_prefetch(Bb0, 0, 1);
#pragma unroll
        for (int kk = 0; kk < HQ; kk += 32) {
            v8h a0 = *(const v8h*)(sHr + kk);
            v8h a1 = *(const v8h*)(sHr + kk + 16);
            v16h Af;
#pragma unroll
            for (int i = 0; i < 8; i++) { Af[i] = a0[i]; Af[i + 8] = a1[i]; }
#pragma unroll
            for (int j = 0; j < 4; j++) {
                const _Float16* Bb = Bb0 + (size_t)j * 16 * HQ + kk;
                v8h b0 = *(const v8h*)(Bb);
                v8h b1 = *(const v8h*)(Bb + 16);
                v16h Bf;
#pragma unroll
                for (int i = 0; i < 8; i++) { Bf[i] = b0[i]; Bf[i + 8] = b1[i]; }
                acc[j] = __builtin_amdgcn_wmma_f32_16x16x32_f16(
                    false, Af, false, Bf, (short)0, acc[j], false, false);
            }
        }
    }

    // ---- store (+bias). C layout: elem r -> M = mt*16 + hi*8 + r, N = ml.
    const int m0 = mt * 16 + hi * 8;
#pragma unroll
    for (int j = 0; j < 4; j++) {
        const int n = (nt0 + j) * 16 + ml;
        const float bi = bias[dir * GQ + n];
        float* gp = gates + (size_t)dir * BQ * GQ + (size_t)m0 * GQ + n;
#pragma unroll
        for (int r = 0; r < 8; r++)
            gp[(size_t)r * GQ] = acc[j][r] + bi;
    }
}

// ---------------------------------------------------------------------------
// LSTM cell pointwise
__global__ void k_lstm_point(const float* __restrict__ gates,
                             float* __restrict__ cws, _Float16* __restrict__ h16,
                             const float* __restrict__ resid, float* __restrict__ out,
                             int T, int s)
{
    int idx = blockIdx.x * blockDim.x + threadIdx.x;
    if (idx >= 2 * BQ * HQ) return;
    int dir  = idx >> 16;
    int u    = idx & 0xFFFF;
    int b    = u >> 8;
    int unit = u & 255;
    int t    = dir ? (T - 1 - s) : s;

    const float* g = gates + (size_t)dir * BQ * GQ + (size_t)b * GQ;
    float gi = g[unit];
    float gf = g[HQ + unit];
    float gg = g[2 * HQ + unit];
    float go = g[3 * HQ + unit];

    float c  = cws[idx];
    float si = 1.f / (1.f + __expf(-gi));
    float sf = 1.f / (1.f + __expf(-gf));
    float so = 1.f / (1.f + __expf(-go));
    c = sf * c + si * tanhf(gg);
    float h = so * tanhf(c);

    cws[idx] = c;
    h16[idx] = (_Float16)h;

    size_t o = ((size_t)b * T + t) * (2 * HQ) + (size_t)dir * HQ + unit;
    out[o] = h + (resid ? resid[o] : 0.f);
}

// ---------------------------------------------------------------------------
// WMMA time-conv (first conv of each head, Cin=512, tanh fused).
__global__ __launch_bounds__(256) void k_conv_wmma(
    const _Float16* __restrict__ Xc,   // (B*T, 512) f16
    const _Float16* __restrict__ Wc,   // (ks, Cout, 512) f16
    const float* __restrict__ bias,
    float* __restrict__ out,           // (B*T, Cout) f32
    int T, int Cout, int ks)
{
    const int wave = threadIdx.x >> 5, lane = threadIdx.x & 31;
    const int tileId = blockIdx.x * 8 + wave;
    const int Mt = (256 * T) >> 4;
    const int mt = tileId % Mt;
    const int nt = tileId / Mt;
    const int ml = lane & 15, hi = lane >> 4;
    const int pad = ks >> 1;

    const int m  = mt * 16 + ml;        // this lane's A row (= output row)
    const int tt = m % T;

    v8f acc = {};
    for (int dk = 0; dk < ks; dk++) {
        const int ts = tt + dk - pad;
        const bool valid = (ts >= 0) && (ts < T);
        const _Float16* Bb = Wc + ((size_t)dk * Cout + nt * 16 + ml) * 512 + hi * 8;
        for (int kk = 0; kk < 512; kk += 32) {
            v8h a0 = {}, a1 = {};
            if (valid) {
                const _Float16* Ab = Xc + (size_t)(m + dk - pad) * 512 + hi * 8 + kk;
                a0 = *(const v8h*)(Ab);
                a1 = *(const v8h*)(Ab + 16);
            }
            v8h b0 = *(const v8h*)(Bb + kk);
            v8h b1 = *(const v8h*)(Bb + kk + 16);
            v16h Af, Bf;
#pragma unroll
            for (int i = 0; i < 8; i++) {
                Af[i] = a0[i]; Af[i + 8] = a1[i];
                Bf[i] = b0[i]; Bf[i + 8] = b1[i];
            }
            acc = __builtin_amdgcn_wmma_f32_16x16x32_f16(
                false, Af, false, Bf, (short)0, acc, false, false);
        }
    }
    const int n = nt * 16 + ml;
    const float bi = bias[n];
    float* op = out + (size_t)(mt * 16 + hi * 8) * Cout + n;
#pragma unroll
    for (int r = 0; r < 8; r++)
        op[(size_t)r * Cout] = tanhf(acc[r] + bi);
}

// ---------------------------------------------------------------------------
// Naive 1-D conv for the small tail convs, layout (B,T,C), weights (O,I,K).
__global__ void k_conv1d(const float* __restrict__ in, int T, int Cin,
                         const float* __restrict__ w, const float* __restrict__ bias,
                         int Cout, int ks, float* __restrict__ out, int tnh)
{
    int idx = blockIdx.x * blockDim.x + threadIdx.x;
    int n = 256 * T * Cout;
    if (idx >= n) return;
    int o = idx % Cout;
    int t = (idx / Cout) % T;
    int b = idx / (Cout * T);
    int pad = ks >> 1;
    float acc = bias[o];
    for (int dk = 0; dk < ks; dk++) {
        int tt = t + dk - pad;
        if (tt < 0 || tt >= T) continue;
        const float* ip = in + ((size_t)b * T + tt) * Cin;
        const float* wp = w + (size_t)o * Cin * ks + dk;
        for (int ci = 0; ci < Cin; ci++)
            acc += ip[ci] * wp[(size_t)ci * ks];
    }
    out[idx] = tnh ? tanhf(acc) : acc;
}

__global__ void k_out4(const float* __restrict__ a, const float* __restrict__ b2,
                       float* __restrict__ o, int n)
{
    int idx = blockIdx.x * blockDim.x + threadIdx.x;
    if (idx < n) o[idx] = a[idx] + b2[idx];
}

// ---------------------------------------------------------------------------
extern "C" void kernel_launch(void* const* d_in, const int* in_sizes, int n_in,
                              void* d_out, int out_size, void* d_ws, size_t ws_size,
                              hipStream_t stream)
{
    (void)in_sizes; (void)n_in; (void)out_size; (void)ws_size;

    const float* x1    = (const float*)d_in[0];
    const float* x3    = (const float*)d_in[1];
    const float* enc_h = (const float*)d_in[2];
    const float* enc_c = (const float*)d_in[3];
    float* out = (float*)d_out;

    char* base = (char*)d_ws;
    size_t off = 0;
    auto alloc = [&](size_t bytes) -> void* {
        void* p = base + off;
        off = (off + bytes + 255) & ~(size_t)255;
        return p;
    };

    float*     A        = (float*)alloc((size_t)256 * 96 * 512 * 4);
    float*     Bf       = (float*)alloc((size_t)256 * 96 * 512 * 4);
    float*     Cf       = (float*)alloc((size_t)256 * 96 * 512 * 4);
    float*     convtmp  = Cf;                      // alias (disjoint lifetime)
    float*     convtmp2 = (float*)alloc((size_t)256 * 96 * 64 * 4);
    _Float16*  X16      = (_Float16*)alloc((size_t)256 * 96 * 512 * 2);
    _Float16*  Wc16     = (_Float16*)alloc((size_t)5 * 256 * 512 * 2);
    float*     gates    = (float*)alloc((size_t)2 * BQ * GQ * 4);
    _Float16*  h16ws    = (_Float16*)alloc((size_t)2 * BQ * HQ * 2);
    float*     cws      = (float*)alloc((size_t)2 * BQ * HQ * 4);
    float*     t04      = (float*)alloc((size_t)256 * 96 * 4);

    struct LL { int whh, wih, bi, din, dpad, stage, subl; };
    const LL L[8] = {
        {26, 27, 28,   2,  32, 0, 0},   // l00
        {29, 30, 31, 512, 512, 1, 0},   // l01
        {32, 33, 34, 512, 512, 3, 0},   // l03[0]
        {35, 36, 37, 512, 512, 3, 1},   // l03[1]
        {38, 39, 40, 512, 512, 4, 0},   // l04[0]
        {41, 42, 43, 512, 512, 4, 1},   // l04[1]
        {44, 45, 46,   3,  32, 5, 0},   // l05[0]
        {47, 48, 49, 512, 512, 5, 1},   // l05[1]
    };

    _Float16* Wih16[8];
    _Float16* Whh16[8];
    for (int i = 0; i < 8; i++) {
        Whh16[i] = (_Float16*)alloc((size_t)2048 * 256 * 2);
        Wih16[i] = (_Float16*)alloc((size_t)2048 * L[i].dpad * 2);
        int nW = 2048 * 256;
        k_padpack<<<(nW + 255) / 256, 256, 0, stream>>>(
            (const float*)d_in[L[i].whh], Whh16[i], nW, 256, 256);
        int nI = 2048 * L[i].dpad;
        k_padpack<<<(nI + 255) / 256, 256, 0, stream>>>(
            (const float*)d_in[L[i].wih], Wih16[i], nI, L[i].din, L[i].dpad);
    }

    auto run_lstm = [&](int li, int T, const float* resid, float* outp) {
        const LL& l = L[li];
        const float* h0 = enc_h + ((size_t)l.stage * 4 + 2 * l.subl) * BQ * HQ;
        const float* c0 = enc_c + ((size_t)l.stage * 4 + 2 * l.subl) * BQ * HQ;
        k_init_state<<<512, 256, 0, stream>>>(h0, c0, h16ws, cws);
        for (int s = 0; s < T; s++) {
            k_lstm_gemm<<<dim3(16, 2, 2), 256, 0, stream>>>(
                X16, T * l.dpad, l.dpad, Wih16[li], Whh16[li],
                (const float*)d_in[l.bi], h16ws, gates, T, s);
            k_lstm_point<<<512, 256, 0, stream>>>(gates, cws, h16ws, resid, outp, T, s);
        }
    };
    auto pack = [&](const float* src, int rows, int din, int dpad) {
        int n = rows * dpad;
        k_padpack<<<(n + 255) / 256, 256, 0, stream>>>(src, X16, n, din, dpad);
    };
    auto conv_wmma = [&](const float* in, int T, int wi, int bi, int Cout, int ks,
                         float* o) {
        pack(in, 256 * T, 512, 512);
        int nw = ks * Cout * 512;
        k_convw_pack<<<(nw + 255) / 256, 256, 0, stream>>>(
            (const float*)d_in[wi], Wc16, Cout, 512, ks, nw);
        int tiles = ((256 * T) >> 4) * (Cout >> 4);
        k_conv_wmma<<<tiles / 8, 256, 0, stream>>>(
            X16, Wc16, (const float*)d_in[bi], o, T, Cout, ks);
    };
    auto conv = [&](const float* in, int T, int Cin, int wi, int bi,
                    int Cout, int ks, float* o, int tnh) {
        int n = 256 * T * Cout;
        k_conv1d<<<(n + 255) / 256, 256, 0, stream>>>(
            in, T, Cin, (const float*)d_in[wi], (const float*)d_in[bi], Cout, ks, o, tnh);
    };
    auto gather = [&](const float* src, int Tin, int Tout, int C, int mul, int dv,
                      float* dst) {
        int n = 256 * Tout * C;
        k_gather_f32<<<(n + 255) / 256, 256, 0, stream>>>(src, dst, Tin, Tout, C, mul, dv, n);
    };

    // ---- l00
    gather(x1, 96, 8, 2, 12, 1, A);
    pack(A, 256 * 8, 2, 32);
    run_lstm(0, 8, nullptr, Bf);
    conv_wmma(Bf, 8, 5, 4, 64, 3, convtmp);
    conv(convtmp, 8, 64, 7, 6, 1, 5, out + 0, 0);

    // ---- l01
    gather(Bf, 8, 12, 512, 1, 4, A);
    pack(A, 256 * 12, 512, 512);
    run_lstm(1, 12, A, Bf);
    conv_wmma(Bf, 12, 9, 8, 64, 3, convtmp);
    conv(convtmp, 12, 64, 11, 10, 1, 5, out + 2048, 0);

    // ---- l03 (2 sub-layers)
    gather(Bf, 12, 36, 512, 1, 6, A);
    pack(A, 256 * 36, 512, 512);
    run_lstm(2, 36, nullptr, Cf);
    pack(Cf, 256 * 36, 512, 512);
    run_lstm(3, 36, A, Bf);
    conv_wmma(Bf, 36, 17, 16, 128, 3, convtmp);
    conv(convtmp, 36, 128, 19, 18, 64, 5, convtmp2, 1);
    conv(convtmp2, 36, 64, 21, 20, 1, 7, out + 29696, 0);

    // ---- l04 (2 sub-layers)
    gather(Bf, 36, 96, 512, 1, 3, A);
    pack(A, 256 * 96, 512, 512);
    run_lstm(4, 96, nullptr, Cf);
    pack(Cf, 256 * 96, 512, 512);
    run_lstm(5, 96, A, Bf);
    conv_wmma(Bf, 96, 23, 22, 256, 5, convtmp);
    conv(convtmp, 96, 256, 25, 24, 1, 7, t04, 0);

    // ---- l05
    {
        int n = 256 * 96 * 3;
        k_build_in5<<<(n + 255) / 256, 256, 0, stream>>>(x1, x3, A, n);
    }
    pack(A, 256 * 96, 3, 32);
    run_lstm(6, 96, nullptr, Cf);
    pack(Cf, 256 * 96, 512, 512);
    run_lstm(7, 96, nullptr, Bf);
    conv_wmma(Bf, 96, 13, 12, 256, 5, convtmp);
    conv(convtmp, 96, 256, 15, 14, 1, 7, out + 5120, 0);

    // ---- out4 = out04 + out05
    {
        int n = 256 * 96;
        k_out4<<<(n + 255) / 256, 256, 0, stream>>>(t04, out + 5120, out + 38912, n);
    }
}
